// CrossLevelAttn_42545946034831
// MI455X (gfx1250) — matrix operand order repaired and checked by reading, not compiled, not run on hardware
//
#include <hip/hip_runtime.h>

typedef __attribute__((ext_vector_type(16))) _Float16 v16h;
typedef __attribute__((ext_vector_type(8)))  _Float16 v8h;
typedef __attribute__((ext_vector_type(8)))  float    v8f;
typedef __attribute__((ext_vector_type(4)))  float    v4f;

// Problem constants (from reference)
constexpr int BB  = 2;
constexpr int HH  = 16;
constexpr int TT  = 2048;
constexpr int DH  = 64;
constexpr int CC  = HH * DH;   // 1024
constexpr int TL1 = 512;
constexpr int TL2 = 128;

constexpr int QB   = 128;  // query rows per block (8 waves x 16)
constexpr int KT   = 64;   // key tile per iteration (4 x 16-key WMMA column tiles)
constexpr int KPAD = 72;   // K tile LDS row stride (halves): 144B rows, 16B aligned
constexpr int VPAD = 72;   // Vt LDS row stride (halves)
constexpr int PPAD = 72;   // P LDS row stride (halves)

// Raw v_exp_f32 (we only need x <= 0, flush-to-zero is fine).
#if __has_builtin(__builtin_amdgcn_exp2f)
#define EXP2F(x) __builtin_amdgcn_exp2f(x)
#else
#define EXP2F(x) exp2f(x)
#endif

static __device__ __forceinline__ v8f wmma16(v16h a, v16h b, v8f c) {
  // D = A(16x32 f16) * B(32x16 f16) + C(16x16 f32)
  return __builtin_amdgcn_wmma_f32_16x16x32_f16(
      /*neg_a=*/false, a, /*neg_b=*/false, b,
      /*c_mod=*/(short)0, c, /*reuse_a=*/false, /*reuse_b=*/false);
}

// Build a v16h fragment from two 16B-aligned 8-half LDS chunks.
static __device__ __forceinline__ v16h lds_frag(const _Float16* p0, const _Float16* p1) {
  v8h lo = *(const v8h*)p0;
  v8h hi = *(const v8h*)p1;
  v16h r = {};
#pragma unroll
  for (int e = 0; e < 8; ++e) { r[e] = lo[e]; r[e + 8] = hi[e]; }
  return r;
}

// 16-lane butterfly max (rows of 16 within wave32).
#if __has_builtin(__builtin_amdgcn_permlane16)
static __device__ __forceinline__ float lane16_xor(float x, unsigned lo, unsigned hi) {
  unsigned xi = __float_as_uint(x);
  unsigned r = __builtin_amdgcn_permlane16(xi, xi, lo, hi, false, false);
  return __uint_as_float(r);
}
static __device__ __forceinline__ float redmax16(float x) {
  x = fmaxf(x, lane16_xor(x, 0x67452301u, 0xEFCDAB89u));  // xor 1
  x = fmaxf(x, lane16_xor(x, 0x54761032u, 0xDCFE98BAu));  // xor 2
  x = fmaxf(x, lane16_xor(x, 0x32107654u, 0xBA98FEDCu));  // xor 4
  x = fmaxf(x, lane16_xor(x, 0xFEDCBA98u, 0x76543210u));  // xor 8
  return x;
}
#else
static __device__ __forceinline__ float redmax16(float x) {
#pragma unroll
  for (int xm = 1; xm < 16; xm <<= 1) x = fmaxf(x, __shfl_xor(x, xm, 32));
  return x;
}
#endif

__global__ __launch_bounds__(256)
void cross_level_attn_wmma(const float* __restrict__ Q,
                           const float* __restrict__ K0, const float* __restrict__ V0,
                           const float* __restrict__ K1, const float* __restrict__ V1,
                           const float* __restrict__ K2, const float* __restrict__ V2,
                           const float* __restrict__ LW,
                           float* __restrict__ Out) {
  // Double-buffered K/V staging + per-wave P scratch: 54 KB total.
  __shared__ __align__(16) _Float16 ldsK[2][KT * KPAD];     // 2 x 9 KB
  __shared__ __align__(16) _Float16 ldsVt[2][DH * VPAD];    // 2 x 9 KB
  __shared__ __align__(16) _Float16 ldsP[8 * 16 * PPAD];    // 18 KB

  constexpr int NQ = TT / QB;  // 16 q-blocks per (b,h)
  const int blk  = blockIdx.x;
  const int qblk = blk % NQ;
  const int bh   = blk / NQ;
  const int h    = bh % HH;
  const int b    = bh / HH;

  const int tid  = threadIdx.x;
  const int wave = tid >> 5;
  const int lane = tid & 31;
  const int ln   = lane & 15;   // column / M index within 16
  const int hi   = lane >> 4;   // half-wave select
  const int g0   = hi * 8;      // A-layout K-group base
  const int kb16 = hi * 16;     // unused; B fragments use g0 too

  const int qb = qblk * QB;
  const int q0 = qb + wave * 16;
  // wave-uniform copy for scalar branching (no EXEC save/restore)
  const int q0u = __builtin_amdgcn_readfirstlane(q0);

  // softmax(level_w) -> per-level mixing weights
  const float w0 = LW[0], w1 = LW[1], w2 = LW[2];
  const float wm = fmaxf(w0, fmaxf(w1, w2));
  const float ew0 = __expf(w0 - wm), ew1 = __expf(w1 - wm), ew2 = __expf(w2 - wm);
  const float winv = 1.f / (ew0 + ew1 + ew2);
  const float wlv[3] = {ew0 * winv, ew1 * winv, ew2 * winv};

  // Q A-fragments with DH^-0.5 * log2(e) folded in:
  // S then arrives already in exp2 domain -> softmax is pure sub+exp2.
  const float qscale = 0.125f * 1.44269504088896f;
  const float* Qp = Q + (((size_t)(b * HH + h)) * TT + (size_t)(q0 + ln)) * DH;
  v16h aq[2] = {};
#pragma unroll
  for (int c = 0; c < 2; ++c) {
#pragma unroll
    for (int e = 0; e < 8; ++e) {
      aq[c][e]     = (_Float16)(Qp[c * 32 + g0 + e] * qscale);
      aq[c][e + 8] = (_Float16)(Qp[c * 32 + g0 + 16 + e] * qscale);
    }
  }

  // all-ones B fragment: rowsum(P) = P x ones via the matrix core
  v16h bones = {};
#pragma unroll
  for (int e = 0; e < 16; ++e) bones[e] = (_Float16)1.0f;

  const float* Kl[3] = {K0, K1, K2};
  const float* Vl[3] = {V0, V1, V2};
  const int    Tl[3] = {TT, TL1, TL2};

  v8f outacc[4] = {};

  // staging map: 256 threads cover 64 keys x 64 d, 16 contiguous floats each.
  const int srow = tid & 63;
  const int sd   = (tid >> 6) * 16;

  for (int lvl = 0; lvl < 3; ++lvl) {
    const float* Kg = Kl[lvl] + ((size_t)b * Tl[lvl]) * CC + (size_t)h * DH;
    const float* Vg = Vl[lvl] + ((size_t)b * Tl[lvl]) * CC + (size_t)h * DH;
    const int kend = (lvl == 0) ? (qb + QB) : Tl[lvl];  // causal early exit on level 0

    // stage tile kt2 into buffer bufsel; prefetch one more tile ahead
    auto stage = [&](int kt2, int bufsel) {
      const float* kp = Kg + (size_t)(kt2 + srow) * CC + sd;
      const float* vp = Vg + (size_t)(kt2 + srow) * CC + sd;
      v4f kf[4], vf[4];
#pragma unroll
      for (int u = 0; u < 4; ++u) { kf[u] = ((const v4f*)kp)[u]; vf[u] = ((const v4f*)vp)[u]; }
      _Float16* dK = ldsK[bufsel];
      _Float16* dV = ldsVt[bufsel];
#pragma unroll
      for (int u = 0; u < 4; ++u)
#pragma unroll
        for (int i = 0; i < 4; ++i) {
          dK[srow * KPAD + sd + u * 4 + i]    = (_Float16)kf[u][i];
          dV[(sd + u * 4 + i) * VPAD + srow]  = (_Float16)vf[u][i];
        }
      if (kt2 + KT < kend) {  // global_prefetch_b8, two tiles ahead of compute
        __builtin_prefetch(kp + (size_t)KT * CC, 0, 1);
        __builtin_prefetch(vp + (size_t)KT * CC, 0, 1);
      }
    };

    v8f o[4] = {};
    float mrow[8], lrow[8];
#pragma unroll
    for (int r = 0; r < 8; ++r) { mrow[r] = -1e30f; lrow[r] = 0.f; }

    __syncthreads();   // previous level's last-tile reads complete
    stage(0, 0);       // prologue: fill buffer 0

    int cur = 0;
    for (int kt = 0; kt < kend; kt += KT) {
      __syncthreads();  // staged tile visible; prior compute reads done

      // stage next tile into the other buffer while this tile computes
      if (kt + KT < kend) stage(kt + KT, cur ^ 1);

      // Wave-uniform causal skip (scalar branch): rows end at q0+15.
      if (!(lvl == 0 && kt > q0u + 15)) {
        const _Float16* sK = ldsK[cur];
        const _Float16* sV = ldsVt[cur];

        // ---- S = Q K^T (pre-scaled): 4 x 16-key tiles, d in 2 chunks
        v8f s[4] = {};
#pragma unroll
        for (int c = 0; c < 2; ++c) {
          v16h bk[4];
#pragma unroll
          for (int n = 0; n < 4; ++n) {
            const _Float16* pk = &sK[(n * 16 + ln) * KPAD + c * 32 + g0];
            bk[n] = lds_frag(pk, pk + 16);
          }
#pragma unroll
          for (int n = 0; n < 4; ++n) s[n] = wmma16(aq[c], bk[n], s[n]);
        }

        // Per-element mask only for the tile straddling the diagonal.
        if (lvl == 0 && kt + KT - 1 > q0u) {
#pragma unroll
          for (int n = 0; n < 4; ++n) {
            const int key = kt + n * 16 + ln;
#pragma unroll
            for (int r = 0; r < 8; ++r)
              if (key > q0 + r + 8 * hi) s[n][r] = -1e30f;
          }
        }

        // ---- online softmax in exp2 domain
        float alpha[8];
#pragma unroll
        for (int r = 0; r < 8; ++r) {
          float mx = fmaxf(fmaxf(s[0][r], s[1][r]), fmaxf(s[2][r], s[3][r]));
          mx = redmax16(mx);
          const float mnew = fmaxf(mrow[r], mx);
          alpha[r] = EXP2F(mrow[r] - mnew);
          mrow[r] = mnew;
#pragma unroll
          for (int n = 0; n < 4; ++n) s[n][r] = EXP2F(s[n][r] - mnew);  // P in s[]
        }
#pragma unroll
        for (int j = 0; j < 4; ++j)
#pragma unroll
          for (int r = 0; r < 8; ++r) o[j][r] *= alpha[r];

        // ---- P: D-layout -> A-layout via per-wave LDS round-trip.
        // Same-wave DS ops execute in order,
        // so the cross-lane RAW needs no explicit drain; the compiler inserts
        // the register-dependency waits for the loads below.
        _Float16* Pw = &ldsP[wave * 16 * PPAD];
#pragma unroll
        for (int r = 0; r < 8; ++r) {
          const int m = r + 8 * hi;
#pragma unroll
          for (int n = 0; n < 4; ++n)
            Pw[m * PPAD + n * 16 + ln] = (_Float16)s[n][r];
        }
        v16h ap0 = lds_frag(&Pw[ln * PPAD + g0],      &Pw[ln * PPAD + g0 + 16]);
        v16h ap1 = lds_frag(&Pw[ln * PPAD + 32 + g0], &Pw[ln * PPAD + 32 + g0 + 16]);

        // ---- row sums on the matrix core: rs = P x ones
        v8f rsv = {};
        rsv = wmma16(ap0, bones, rsv);
        rsv = wmma16(ap1, bones, rsv);
#pragma unroll
        for (int r = 0; r < 8; ++r) lrow[r] = lrow[r] * alpha[r] + rsv[r];

        // ---- O += P * V : 4 d-column tiles, keys in 2 chunks of 32
        v16h bv[4];
#pragma unroll
        for (int j = 0; j < 4; ++j) {
          const _Float16* pv0 = &sV[(j * 16 + ln) * VPAD + g0];
          bv[j] = lds_frag(pv0, pv0 + 16);
        }
#pragma unroll
        for (int j = 0; j < 4; ++j) o[j] = wmma16(ap0, bv[j], o[j]);
#pragma unroll
        for (int j = 0; j < 4; ++j) {
          const _Float16* pv1 = &sV[(j * 16 + ln) * VPAD + 32 + g0];
          bv[j] = lds_frag(pv1, pv1 + 16);
        }
#pragma unroll
        for (int j = 0; j < 4; ++j) o[j] = wmma16(ap1, bv[j], o[j]);
      }  // active

      cur ^= 1;
    }  // key tiles

    const float wl = wlv[lvl];
#pragma unroll
    for (int r = 0; r < 8; ++r) {
      const float f = wl / lrow[r];
#pragma unroll
      for (int j = 0; j < 4; ++j) outacc[j][r] += f * o[j][r];
    }
  }  // levels

  // out[b, t, h*DH + d]
#pragma unroll
  for (int j = 0; j < 4; ++j) {
#pragma unroll
    for (int r = 0; r < 8; ++r) {
      const int qrow = q0 + r + 8 * hi;
      Out[((size_t)b * TT + qrow) * CC + h * DH + j * 16 + ln] = outacc[j][r];
    }
  }
}

extern "C" void kernel_launch(void* const* d_in, const int* in_sizes, int n_in,
                              void* d_out, int out_size, void* d_ws, size_t ws_size,
                              hipStream_t stream) {
  (void)in_sizes; (void)n_in; (void)out_size; (void)d_ws; (void)ws_size;
  const float* Q  = (const float*)d_in[0];
  const float* K0 = (const float*)d_in[1];
  const float* V0 = (const float*)d_in[2];
  const float* K1 = (const float*)d_in[3];
  const float* V1 = (const float*)d_in[4];
  const float* K2 = (const float*)d_in[5];
  const float* V2 = (const float*)d_in[6];
  const float* LW = (const float*)d_in[7];
  float* Out = (float*)d_out;

  dim3 grid(BB * HH * (TT / QB));  // 512 blocks
  dim3 block(256);                 // 8 waves (wave32)
  hipLaunchKernelGGL(cross_level_attn_wmma, grid, block, 0, stream,
                     Q, K0, V0, K1, V1, K2, V2, LW, Out);
}
